// BrickedAttention_82386062672334
// MI455X (gfx1250) — matrix operand order repaired and checked by reading, not compiled, not run on hardware
//
#include <hip/hip_runtime.h>

// ---------------------------------------------------------------------------
// BrickedAttention for MI455X (gfx1250): bf16 WMMA GEMMs (async-LDS double-
// buffered) + flash window attention.
// ---------------------------------------------------------------------------

typedef __attribute__((ext_vector_type(16))) __bf16 v16bf;
typedef __attribute__((ext_vector_type(8)))  __bf16 v8bf;
typedef __attribute__((ext_vector_type(8)))  float  v8f;

constexpr int NB   = 4;      // batch
constexpr int LSEQ = 8192;   // sequence length
constexpr int LP   = 8448;   // padded length (WINDOW multiple + 256)
constexpr int EMBD = 1024;
constexpr int NH   = 16;
constexpr int HD   = 64;
constexpr int WIN  = 256;
constexpr int ROWS = NB * LP;      // 33792 (divisible by 128)
constexpr int NW0  = LP / WIN;     // 33 windows, pass 0
constexpr int NW1  = NW0 + 1;      // 34 windows, pass 1 (shift 128)

// ------------------------------ small helpers ------------------------------

static __device__ inline v8f zero_v8f() {
  v8f z;
#pragma unroll
  for (int i = 0; i < 8; ++i) z[i] = 0.f;
  return z;
}
static __device__ inline v8bf zero_v8bf() {
  v8bf z;
#pragma unroll
  for (int i = 0; i < 8; ++i) z[i] = (__bf16)0.f;
  return z;
}
static __device__ inline v16bf zero_v16bf() {
  v16bf z;
#pragma unroll
  for (int i = 0; i < 16; ++i) z[i] = (__bf16)0.f;
  return z;
}

// A-fragment (16x32 bf16, M x K): lane holds row m = lane%16.
// elements 0-7  -> k = half*8 + i        (16B contiguous)
// elements 8-15 -> k = 16 + half*8 + i-8 (16B contiguous, +16 elems)
static __device__ inline v16bf frag_a(const __bf16* rowp, int hf) {
  v8bf lo = *(const v8bf*)(rowp + hf * 8);
  v8bf hi = *(const v8bf*)(rowp + hf * 8 + 16);
  return __builtin_shufflevector(lo, hi, 0, 1, 2, 3, 4, 5, 6, 7,
                                 8, 9, 10, 11, 12, 13, 14, 15);
}
// B-fragment (32x16 bf16, K x N): lane holds column n = lane%16,
// elements 0-7 -> k = half*8 + i, elements 8-15 -> k = 16 + half*8 + i-8.
static __device__ inline v16bf frag_b(const __bf16* colp, int hf) {
  v8bf lo = *(const v8bf*)(colp + hf * 8);
  v8bf hi = *(const v8bf*)(colp + hf * 8 + 16);
  return __builtin_shufflevector(lo, hi, 0, 1, 2, 3, 4, 5, 6, 7,
                                 8, 9, 10, 11, 12, 13, 14, 15);
}
static __device__ inline v8f wmma_bf16(v16bf a, v16bf b, v8f c) {
  return __builtin_amdgcn_wmma_f32_16x16x32_bf16(false, a, false, b,
                                                 (short)0, c, false, false);
}
static __device__ inline float block_sum(float v, float* red, int tid) {
  red[tid] = v;
  __syncthreads();
  for (int s = 128; s > 0; s >>= 1) {
    if (tid < s) red[tid] += red[tid + s];
    __syncthreads();
  }
  float r = red[0];
  __syncthreads();
  return r;
}

// Async global->LDS copy (16B per lane), tracked by ASYNCcnt.
// LDS byte offset = low 32 bits of the generic pointer (flat LDS aperture:
// CDNA5 ISA aperture table).
static __device__ inline void async_ld_b128(const void* gp, void* lp) {
  unsigned loff = (unsigned)(unsigned long long)lp;
  asm volatile("global_load_async_to_lds_b128 %0, %1, off"
               :: "v"(loff), "v"(gp)
               : "memory");
}
template <int N>
static __device__ inline void s_wait_async() {
  asm volatile("s_wait_asynccnt %0" :: "i"(N) : "memory");
}

// --------------------------- prep: bf16 convert ----------------------------

__global__ __launch_bounds__(256) void convx_kernel(const float* __restrict__ x,
                                                    __bf16* __restrict__ xp) {
  size_t idx = (size_t)blockIdx.x * 256 + threadIdx.x;  // over ROWS*EMBD
  int c = (int)(idx % EMBD);
  size_t row = idx / EMBD;
  int b = (int)(row / LP), ib = (int)(row % LP);
  float v = (ib < LSEQ) ? x[((size_t)b * LSEQ + ib) * EMBD + c] : 0.f;
  xp[idx] = (__bf16)v;
}

__global__ __launch_bounds__(256) void transpose_w_kernel(const float* __restrict__ W,
                                                          __bf16* __restrict__ Wt) {
  size_t idx = (size_t)blockIdx.x * 256 + threadIdx.x;  // idx = n*EMBD + k
  int k = (int)(idx % EMBD);
  int n = (int)(idx / EMBD);
  Wt[idx] = (__bf16)W[(size_t)k * EMBD + n];
}

// ------------------------------- bf16 GEMM ---------------------------------
// C[M,N] = A[M,K](bf16 row-major) * Bt[N,K](bf16, pre-transposed weights)
// Block tile 128x128, BK=64 (two WMMA K steps per stage). 8 waves = 2x4 wave
// grid, each wave computes 64x32 via 4x2 WMMA accumulators. Tiles are staged
// into double-buffered LDS with GLOBAL_LOAD_ASYNC_TO_LDS_B128 and pipelined:
// issue stage t+1, s_wait_asynccnt for stage t (in-order completion), barrier,
// 16 WMMAs, barrier.

template <bool OUTBF>
__global__ __launch_bounds__(256) void gemm_bf16_kernel(
    const __bf16* __restrict__ A, const __bf16* __restrict__ Bt,
    void* __restrict__ Cout, int M, int N, int K) {
  constexpr int BM = 128, BN = 128, BK = 64, LDT = 72;  // 72*2B=144B, 16B-mult
  __shared__ __bf16 Asb[2][BM * LDT];
  __shared__ __bf16 Bsb[2][BN * LDT];

  const int tid = threadIdx.x;
  const int lane = tid & 31, hf = lane >> 4, ln = lane & 15;
  const int wv = tid >> 5, wrow = wv >> 2, wcol = wv & 3;
  const int m0 = blockIdx.y * BM, n0 = blockIdx.x * BN;

  // Issue one stage of async tile copies: 128 rows x 64 cols bf16 per matrix,
  // 8 chunks of 16B per row; 256 threads x 4 chunks x 2 matrices = 8 async
  // instructions per wave -> ASYNCcnt threshold 8 for one in-flight stage.
  auto stage = [&](int buf, int k0) {
#pragma unroll
    for (int i = 0; i < 4; ++i) {
      int c = tid + i * 256;           // 1024 chunks of 8 bf16
      int row = c >> 3, seg = c & 7;   // 8 chunks per 64-wide row
      async_ld_b128(A + (size_t)(m0 + row) * K + k0 + seg * 8,
                    Asb[buf] + row * LDT + seg * 8);
      async_ld_b128(Bt + (size_t)(n0 + row) * K + k0 + seg * 8,
                    Bsb[buf] + row * LDT + seg * 8);
    }
  };

  v8f acc[4][2];
#pragma unroll
  for (int i = 0; i < 4; ++i)
#pragma unroll
    for (int j = 0; j < 2; ++j) acc[i][j] = zero_v8f();

  const int T = K / BK;  // 16 stages
  stage(0, 0);
  for (int t = 0; t < T; ++t) {
    if (t + 1 < T) {
      stage((t + 1) & 1, (t + 1) * BK);
      s_wait_async<8>();   // stage t done (in-order); t+1 still in flight
    } else {
      s_wait_async<0>();
    }
    __syncthreads();

    const __bf16* As = Asb[t & 1];
    const __bf16* Bs = Bsb[t & 1];
#pragma unroll
    for (int dc = 0; dc < 2; ++dc) {  // two WMMA K-chunks per stage
      v16bf af[4], bfg[2];
#pragma unroll
      for (int mf = 0; mf < 4; ++mf)
        af[mf] = frag_a(As + (wrow * 64 + mf * 16 + ln) * LDT + dc * 32, hf);
#pragma unroll
      for (int nf = 0; nf < 2; ++nf)
        bfg[nf] = frag_b(Bs + (wcol * 32 + nf * 16 + ln) * LDT + dc * 32, hf);
#pragma unroll
      for (int mf = 0; mf < 4; ++mf)
#pragma unroll
        for (int nf = 0; nf < 2; ++nf)
          acc[mf][nf] = wmma_bf16(af[mf], bfg[nf], acc[mf][nf]);
    }
    __syncthreads();  // reads done before this buffer is overwritten at t+2
  }

#pragma unroll
  for (int mf = 0; mf < 4; ++mf)
#pragma unroll
    for (int nf = 0; nf < 2; ++nf)
#pragma unroll
      for (int r = 0; r < 8; ++r) {
        size_t grow = (size_t)m0 + wrow * 64 + mf * 16 + r + hf * 8;
        size_t gcol = (size_t)n0 + wcol * 32 + nf * 16 + ln;
        if (OUTBF)
          ((__bf16*)Cout)[grow * N + gcol] = (__bf16)acc[mf][nf][r];
        else
          ((float*)Cout)[grow * N + gcol] = acc[mf][nf][r];
      }
}

// --------------------------- flash window attention ------------------------
// One block = (head, window, batch). 256 thr = 8 waves; each wave owns 32
// query rows. Computes S^T = K*Q^T so softmax state is lane-local per query,
// online-softmax over 8 chunks of 32 keys, P*V accumulated in fp32 frags.
// Stores 0.5*softmax(QK^T/8)V into attnb (PASS0: store, PASS1: read-add).

template <int PASS>
__global__ __launch_bounds__(256) void attn_kernel(
    const __bf16* __restrict__ Q, const __bf16* __restrict__ Kd,
    const __bf16* __restrict__ V, __bf16* __restrict__ attnb) {
  constexpr int KSTR = 72;   // 144B rows: 16B-aligned, bank-skewed
  constexpr int VSTR = 264;  // 528B rows: 16B-aligned, bank-skewed
  __shared__ __bf16 Ks[WIN * KSTR];   // 256 keys x 64 dims
  __shared__ __bf16 Vt[HD * VSTR];    // transposed: 64 dims x 256 keys

  const int h = blockIdx.x, w = blockIdx.y, b = blockIdx.z;
  const int shift = (PASS == 0) ? 0 : (WIN / 2);
  const int wstart = w * WIN - shift;
  const int tid = threadIdx.x;
  const int lane = tid & 31, hf = lane >> 4, ln = lane & 15;
  const int wv = tid >> 5;

  const __bf16* Kbase = Kd + ((size_t)b * LP) * EMBD + h * HD;
  const __bf16* Vbase = V + ((size_t)b * LP) * EMBD + h * HD;
  const __bf16* Qbase = Q + ((size_t)b * LP) * EMBD + h * HD;

  // Stage K (row-major) and V (transposed) into LDS; OOB rows are zero,
  // which exactly matches the reference's zero-padded windows.
#pragma unroll
  for (int i = 0; i < 8; ++i) {
    int c = tid + i * 256;               // 2048 chunks of 8 bf16
    int key = c >> 3, seg = c & 7;
    int grow = wstart + key;
    v8bf kv = zero_v8bf(), vv = zero_v8bf();
    if (grow >= 0 && grow < LP) {
      kv = *(const v8bf*)(Kbase + (size_t)grow * EMBD + seg * 8);
      vv = *(const v8bf*)(Vbase + (size_t)grow * EMBD + seg * 8);
    }
    *(v8bf*)(Ks + key * KSTR + seg * 8) = kv;
#pragma unroll
    for (int j = 0; j < 8; ++j) Vt[(seg * 8 + j) * VSTR + key] = vv[j];
  }
  __syncthreads();

  // Preload Q^T B-fragments: [query tile][d-chunk of 32]
  v16bf qf[2][2];
#pragma unroll
  for (int qt = 0; qt < 2; ++qt)
#pragma unroll
    for (int dc = 0; dc < 2; ++dc) {
      int grow = wstart + wv * 32 + qt * 16 + ln;
      qf[qt][dc] = (grow >= 0 && grow < LP)
                       ? frag_b(Qbase + (size_t)grow * EMBD + dc * 32, hf)
                       : zero_v16bf();
    }

  v8f out[2][4];
#pragma unroll
  for (int qt = 0; qt < 2; ++qt)
#pragma unroll
    for (int dt = 0; dt < 4; ++dt) out[qt][dt] = zero_v8f();
  float mold[2] = {-1e30f, -1e30f};
  float lsum[2] = {0.f, 0.f};

  for (int chunk = 0; chunk < 8; ++chunk) {  // 32 keys per chunk
    // S^T frags: D rows = keys, D cols (lanes) = queries
    v16bf kf[2][2];
#pragma unroll
    for (int kt = 0; kt < 2; ++kt)
#pragma unroll
      for (int dc = 0; dc < 2; ++dc)
        kf[kt][dc] = frag_a(Ks + (chunk * 32 + kt * 16 + ln) * KSTR + dc * 32, hf);
    v8f s[2][2];
#pragma unroll
    for (int kt = 0; kt < 2; ++kt)
#pragma unroll
      for (int qt = 0; qt < 2; ++qt) {
        s[kt][qt] = zero_v8f();
#pragma unroll
        for (int dc = 0; dc < 2; ++dc)
          s[kt][qt] = wmma_bf16(kf[kt][dc], qf[qt][dc], s[kt][qt]);
      }

#pragma unroll
    for (int qt = 0; qt < 2; ++qt) {
      float p[2][8];
      float mloc = -1e30f;
#pragma unroll
      for (int kt = 0; kt < 2; ++kt)
#pragma unroll
        for (int r = 0; r < 8; ++r) {
          float sv = s[kt][qt][r] * 0.125f;  // / sqrt(64)
          p[kt][r] = sv;
          mloc = fmaxf(mloc, sv);
        }
      // merge the two lane-halves of each query row
      float m2 = fmaxf(mloc, __shfl_xor(mloc, 16, 32));
      float mnew = fmaxf(mold[qt], m2);
      float scl = __expf(mold[qt] - mnew);
      mold[qt] = mnew;
      float ls = 0.f;
#pragma unroll
      for (int kt = 0; kt < 2; ++kt)
#pragma unroll
        for (int r = 0; r < 8; ++r) {
          p[kt][r] = __expf(p[kt][r] - mnew);
          ls += p[kt][r];
        }
      lsum[qt] = lsum[qt] * scl + ls;
      // rescale accumulator: row r of D-frag is query r + half*8 -> its
      // scale lives in lane (r + half*8); broadcast with shuffles.
#pragma unroll
      for (int dt = 0; dt < 4; ++dt)
#pragma unroll
        for (int r = 0; r < 8; ++r)
          out[qt][dt][r] *= __shfl(scl, r + hf * 8, 32);
      // Build P A-fragment: elems 0-7 = keys half*8+i (S^T tile kt0),
      // elems 8-15 = keys 16+half*8+i (S^T tile kt1) -- lane-local!
      v16bf pf;
#pragma unroll
      for (int i = 0; i < 8; ++i) {
        pf[i] = (__bf16)p[0][i];
        pf[i + 8] = (__bf16)p[1][i];
      }
#pragma unroll
      for (int dt = 0; dt < 4; ++dt) {
        v16bf vf = frag_b(Vt + (dt * 16 + ln) * VSTR + chunk * 32, hf);
        out[qt][dt] = wmma_bf16(pf, vf, out[qt][dt]);
      }
    }
  }

  // Normalize by softmax denominator and store 0.5 * result.
  __bf16* Abase = attnb + ((size_t)b * LP) * EMBD + h * HD;
#pragma unroll
  for (int qt = 0; qt < 2; ++qt) {
    float lt = lsum[qt] + __shfl_xor(lsum[qt], 16, 32);
    float rinv[8];
#pragma unroll
    for (int r = 0; r < 8; ++r) rinv[r] = 0.5f / __shfl(lt, r + hf * 8, 32);
#pragma unroll
    for (int dt = 0; dt < 4; ++dt)
#pragma unroll
      for (int r = 0; r < 8; ++r) {
        int grow = wstart + wv * 32 + qt * 16 + r + hf * 8;
        if (grow >= 0 && grow < LP) {
          size_t idx = (size_t)grow * EMBD + dt * 16 + ln;
          float val = out[qt][dt][r] * rinv[r];
          if (PASS == 0)
            Abase[idx] = (__bf16)val;
          else
            Abase[idx] = (__bf16)((float)Abase[idx] + val);
        }
      }
  }
}

// --------------------------- fused LN epilogues ----------------------------

__global__ __launch_bounds__(256) void ln1_kernel(
    const float* __restrict__ x, const float* __restrict__ t1,
    const float* __restrict__ g, const float* __restrict__ bb,
    float* __restrict__ mh, __bf16* __restrict__ mhb) {
  __shared__ float red[256];
  const int row = blockIdx.x;
  const int b = row / LP, ib = row % LP;
  const int tid = threadIdx.x;
  float v[4];
#pragma unroll
  for (int j = 0; j < 4; ++j) {
    int c = tid + j * 256;
    float xv = (ib < LSEQ) ? x[((size_t)b * LSEQ + ib) * EMBD + c] : 0.f;
    v[j] = xv + t1[(size_t)row * EMBD + c];
  }
  float mu = block_sum(v[0] + v[1] + v[2] + v[3], red, tid) * (1.f / EMBD);
  float q = 0.f;
#pragma unroll
  for (int j = 0; j < 4; ++j) {
    float d = v[j] - mu;
    q += d * d;
  }
  float var = block_sum(q, red, tid) * (1.f / EMBD);
  float rs = rsqrtf(var + 1e-5f);
#pragma unroll
  for (int j = 0; j < 4; ++j) {
    int c = tid + j * 256;
    float y = (v[j] - mu) * rs * g[c] + bb[c];
    mh[(size_t)row * EMBD + c] = y;
    mhb[(size_t)row * EMBD + c] = (__bf16)y;
  }
}

__global__ __launch_bounds__(256) void ln2_kernel(
    const float* __restrict__ t2, const float* __restrict__ mh,
    const float* __restrict__ bout, const float* __restrict__ g,
    const float* __restrict__ bb, float* __restrict__ outp) {
  __shared__ float red[256];
  const int row = blockIdx.x;  // over NB*LSEQ (pad rows dropped)
  const int b = row / LSEQ, ib = row % LSEQ;
  const size_t prow = (size_t)b * LP + ib;
  const int tid = threadIdx.x;
  float v[4];
#pragma unroll
  for (int j = 0; j < 4; ++j) {
    int c = tid + j * 256;
    v[j] = t2[prow * EMBD + c] + bout[c] + mh[prow * EMBD + c];
  }
  float mu = block_sum(v[0] + v[1] + v[2] + v[3], red, tid) * (1.f / EMBD);
  float q = 0.f;
#pragma unroll
  for (int j = 0; j < 4; ++j) {
    float d = v[j] - mu;
    q += d * d;
  }
  float var = block_sum(q, red, tid) * (1.f / EMBD);
  float rs = rsqrtf(var + 1e-5f);
#pragma unroll
  for (int j = 0; j < 4; ++j) {
    int c = tid + j * 256;
    float y = (v[j] - mu) * rs * g[c] + bb[c];
    outp[(size_t)row * EMBD + c] = fmaxf(y, 0.f);
  }
}

// ------------------------------- launcher ----------------------------------

extern "C" void kernel_launch(void* const* d_in, const int* in_sizes, int n_in,
                              void* d_out, int out_size, void* d_ws,
                              size_t ws_size, hipStream_t stream) {
  (void)in_sizes; (void)n_in; (void)out_size; (void)ws_size;

  const float* x     = (const float*)d_in[0];
  const float* W_q   = (const float*)d_in[1];  // (2,E,E)
  const float* W_k   = (const float*)d_in[2];
  const float* W_v   = (const float*)d_in[3];
  const float* W_o   = (const float*)d_in[4];
  const float* W_out = (const float*)d_in[5];
  const float* b_out = (const float*)d_in[6];
  const float* ln1_g = (const float*)d_in[7];
  const float* ln1_b = (const float*)d_in[8];
  const float* ln2_g = (const float*)d_in[9];
  const float* ln2_b = (const float*)d_in[10];
  float* outp = (float*)d_out;

  // workspace carve-up (256B aligned)
  char* ws = (char*)d_ws;
  size_t off = 0;
  auto alloc = [&](size_t bytes) -> void* {
    void* p = ws + off;
    off = (off + bytes + 255) & ~(size_t)255;
    return p;
  };
  const size_t XPB  = (size_t)ROWS * EMBD * sizeof(__bf16);
  const size_t WTB  = (size_t)EMBD * EMBD * sizeof(__bf16);
  const size_t TMPF = (size_t)ROWS * EMBD * sizeof(float);

  __bf16* xp    = (__bf16*)alloc(XPB);
  __bf16* wqT0  = (__bf16*)alloc(WTB);
  __bf16* wqT1  = (__bf16*)alloc(WTB);
  __bf16* wkT0  = (__bf16*)alloc(WTB);
  __bf16* wkT1  = (__bf16*)alloc(WTB);
  __bf16* wvT0  = (__bf16*)alloc(WTB);
  __bf16* wvT1  = (__bf16*)alloc(WTB);
  __bf16* woT   = (__bf16*)alloc(WTB);
  __bf16* woutT = (__bf16*)alloc(WTB);
  __bf16* qb    = (__bf16*)alloc(XPB);
  __bf16* kb    = (__bf16*)alloc(XPB);
  __bf16* vb    = (__bf16*)alloc(XPB);
  __bf16* attnb = (__bf16*)alloc(XPB);
  float*  tmpf  = (float*)alloc(TMPF);   // reused for both projection GEMMs
  float*  mh    = (float*)alloc(TMPF);
  __bf16* mhb   = (__bf16*)alloc(XPB);

  const int TOTAL = ROWS * EMBD;               // 34,603,008 (256-divisible)
  const int WELEM = EMBD * EMBD;               // 1,048,576
  const dim3 gemm_grid(EMBD / 128, ROWS / 128);  // (8, 264)

  // prep: bf16 x (zero-padded) and transposed bf16 weights
  convx_kernel<<<TOTAL / 256, 256, 0, stream>>>(x, xp);
  transpose_w_kernel<<<WELEM / 256, 256, 0, stream>>>(W_q, wqT0);
  transpose_w_kernel<<<WELEM / 256, 256, 0, stream>>>(W_q + WELEM, wqT1);
  transpose_w_kernel<<<WELEM / 256, 256, 0, stream>>>(W_k, wkT0);
  transpose_w_kernel<<<WELEM / 256, 256, 0, stream>>>(W_k + WELEM, wkT1);
  transpose_w_kernel<<<WELEM / 256, 256, 0, stream>>>(W_v, wvT0);
  transpose_w_kernel<<<WELEM / 256, 256, 0, stream>>>(W_v + WELEM, wvT1);
  transpose_w_kernel<<<WELEM / 256, 256, 0, stream>>>(W_o, woT);
  transpose_w_kernel<<<WELEM / 256, 256, 0, stream>>>(W_out, woutT);

  // pass 0: QKV projections + windowed attention (shift 0)
  gemm_bf16_kernel<true><<<gemm_grid, 256, 0, stream>>>(xp, wqT0, qb, ROWS, EMBD, EMBD);
  gemm_bf16_kernel<true><<<gemm_grid, 256, 0, stream>>>(xp, wkT0, kb, ROWS, EMBD, EMBD);
  gemm_bf16_kernel<true><<<gemm_grid, 256, 0, stream>>>(xp, wvT0, vb, ROWS, EMBD, EMBD);
  attn_kernel<0><<<dim3(NH, NW0, NB), 256, 0, stream>>>(qb, kb, vb, attnb);

  // pass 1: QKV projections + windowed attention (shift 128), accumulate
  gemm_bf16_kernel<true><<<gemm_grid, 256, 0, stream>>>(xp, wqT1, qb, ROWS, EMBD, EMBD);
  gemm_bf16_kernel<true><<<gemm_grid, 256, 0, stream>>>(xp, wkT1, kb, ROWS, EMBD, EMBD);
  gemm_bf16_kernel<true><<<gemm_grid, 256, 0, stream>>>(xp, wvT1, vb, ROWS, EMBD, EMBD);
  attn_kernel<1><<<dim3(NH, NW1, NB), 256, 0, stream>>>(qb, kb, vb, attnb);

  // mh = LN1(x + attn @ W_o)
  gemm_bf16_kernel<false><<<gemm_grid, 256, 0, stream>>>(attnb, woT, tmpf, ROWS, EMBD, EMBD);
  ln1_kernel<<<ROWS, 256, 0, stream>>>(x, tmpf, ln1_g, ln1_b, mh, mhb);

  // out = relu(LN2(mh @ W_out + b_out + mh)), pad rows dropped
  gemm_bf16_kernel<false><<<gemm_grid, 256, 0, stream>>>(mhb, woutT, tmpf, ROWS, EMBD, EMBD);
  ln2_kernel<<<NB * LSEQ, 256, 0, stream>>>(tmpf, mh, b_out, ln2_g, ln2_b, outp);
}
